// Model_69286412419591
// MI455X (gfx1250) — compile-verified, matrix operand order checked by reading, not run on hardware
//
#include <hip/hip_runtime.h>
#include <cstdint>
#include <cstddef>

typedef __attribute__((ext_vector_type(16))) __bf16 v16bf;
typedef __attribute__((ext_vector_type(8)))  float  v8f;

// ---------------- problem constants ----------------
constexpr int Nn = 512, Tt = 600, Dd = 56, Cc = 30, nHh = 5;
constexpr int Ll = 20, Ff = 280, Hh = 256, NSs = 10;
constexpr int Bb = Nn * NSs;           // 5120
constexpr int H3 = 3 * Hh;             // 768
constexpr int LF = Ll * Ff;            // 5600
constexpr float PERT = 10000.0f, LAMB = 1e-3f, EPSBN = 1e-5f;

static __device__ __forceinline__ unsigned short f2bf(float x) {
  unsigned int u = __float_as_uint(x);
  return (unsigned short)((u + 0x7FFFu + ((u >> 16) & 1u)) >> 16);
}
static __device__ __forceinline__ float sigm(float x) { return 1.0f / (1.0f + __expf(-x)); }

struct U8 { unsigned int u[8]; };

// ---------------- WMMA bf16 GEMM ----------------
// C[M,Ncols] = A[M,K](f32) * Bt^T (+Cadd) (+bias)
// Bt is [n][k] row-major bf16, row length ldbt == Kpad (Kpad % 32 == 0, zero padded).
// REQUIREMENTS (all call sites satisfy): M % 128 == 0, lda % 4 == 0,
// A/Bt 16-byte aligned.
// Block: 128 threads = 4 waves. Block tile 128x64. Wave tile 32x64 -> 8 WMMAs/chunk.
constexpr int TSu = 40;  // LDS row stride in ushorts (80B: 16B aligned, bank-spread)

__global__ void __launch_bounds__(128)
gemm_bf16(const float* __restrict__ A, int lda,
          const unsigned short* __restrict__ Bt, int ldbt,
          float* __restrict__ C, int ldc,
          const float* __restrict__ Cadd, int ldca,
          const float* __restrict__ bias,
          int M, int Ncols, int K, int Kpad)
{
  __shared__ unsigned short As[128 * TSu]; // 10.0 KB
  __shared__ unsigned short Bs[64 * TSu];  //  5.0 KB
  const int tid  = threadIdx.x;
  const int lane = tid & 31;
  const int wave = tid >> 5;
  const int lmod = lane & 15;
  const int lhi  = lane >> 4;       // 0 or 1
  const int j0   = lhi << 2;        // dword offset within LDS row: 0 or 4
  const int m0   = blockIdx.x * 128;
  const int n0   = blockIdx.y * 64;

  const v8f vzero = {0.f,0.f,0.f,0.f,0.f,0.f,0.f,0.f};
  v8f acc[2][4];
#pragma unroll
  for (int mi = 0; mi < 2; ++mi)
#pragma unroll
    for (int nt = 0; nt < 4; ++nt) acc[mi][nt] = vzero;

  for (int kk = 0; kk < Kpad; kk += 32) {
    // ---- A tile 128x32 (f32 -> bf16) ----
    if (kk + 32 <= K) {
      // fast path: unguarded 16B loads
#pragma unroll
      for (int i = 0; i < 8; ++i) {
        int idx = tid + i * 128;                // 1024 float4 chunks
        int r = idx >> 3, c4 = (idx & 7) << 2;  // row, col (floats)
        const float4 f = *(const float4*)(A + (size_t)(m0 + r) * lda + kk + c4);
        unsigned int p0 = (unsigned int)f2bf(f.x) | ((unsigned int)f2bf(f.y) << 16);
        unsigned int p1 = (unsigned int)f2bf(f.z) | ((unsigned int)f2bf(f.w) << 16);
        *(uint2*)(As + r * TSu + c4) = make_uint2(p0, p1);
      }
    } else {
      // K tail: branchless clamp + select (only last chunk of K=280 GEMMs)
#pragma unroll
      for (int i = 0; i < 32; ++i) {
        int idx = tid + i * 128;                // 4096 elements
        int r = idx >> 5, c = idx & 31;
        int gk = kk + c;
        int gkc = gk < K ? gk : (K - 1);
        float av = A[(size_t)(m0 + r) * lda + gkc];
        As[r * TSu + c] = f2bf(gk < K ? av : 0.0f);
      }
    }
    // ---- B tile 64x32 (bf16, rows clamped+masked) ----
#pragma unroll
    for (int i = 0; i < 2; ++i) {
      int idx = tid + i * 128;                  // 256 chunks of 8 ushorts
      int r = idx >> 2, c8 = (idx & 3) << 3;
      int gn = n0 + r;
      int gnc = gn < Ncols ? gn : (Ncols - 1);
      uint4 v = *(const uint4*)(Bt + (size_t)gnc * ldbt + kk + c8);
      unsigned int msk = gn < Ncols ? 0xFFFFFFFFu : 0u;
      v.x &= msk; v.y &= msk; v.z &= msk; v.w &= msk;
      *(uint4*)(Bs + r * TSu + c8) = v;
    }
    __syncthreads();

    union FR { U8 b; v16bf v; };
    FR af[2];
#pragma unroll
    for (int mi = 0; mi < 2; ++mi) {
      const unsigned int* p = (const unsigned int*)(As + (wave * 32 + mi * 16 + lmod) * TSu);
#pragma unroll
      for (int i = 0; i < 4; ++i) { af[mi].b.u[i] = p[j0 + i]; af[mi].b.u[4 + i] = p[j0 + 8 + i]; }
    }
#pragma unroll
    for (int nt = 0; nt < 4; ++nt) {
      FR bf_;
      const unsigned int* p = (const unsigned int*)(Bs + (nt * 16 + lmod) * TSu);
#pragma unroll
      for (int i = 0; i < 4; ++i) { bf_.b.u[i] = p[j0 + i]; bf_.b.u[4 + i] = p[j0 + 8 + i]; }
#pragma unroll
      for (int mi = 0; mi < 2; ++mi)
        acc[mi][nt] = __builtin_amdgcn_wmma_f32_16x16x32_bf16(
            false, af[mi].v, false, bf_.v, (short)0, acc[mi][nt], false, false);
    }
    __syncthreads();
  }

  // epilogue (M % 128 == 0 -> no row guard)
  const int cb = n0 + lmod;
#pragma unroll
  for (int mi = 0; mi < 2; ++mi) {
    const int rbase = m0 + wave * 32 + mi * 16 + (lhi << 3);
#pragma unroll
    for (int nt = 0; nt < 4; ++nt) {
      int cc2 = cb + nt * 16;
      if (cc2 < Ncols) {
#pragma unroll
        for (int e = 0; e < 8; ++e) {
          int rr = rbase + e;
          float v = acc[mi][nt][e];
          if (Cadd) v += Cadd[(size_t)rr * ldca + cc2];
          if (bias) v += bias[cc2];
          C[(size_t)rr * ldc + cc2] = v;
        }
      }
    }
  }
}

// ---------------- attention ----------------
__global__ void __launch_bounds__(256)
attn_kernel(const float* __restrict__ data, const float* __restrict__ Wq,
            const float* __restrict__ Wk, float* __restrict__ vals)
{
  __shared__ float X[Cc * Dd];       // 1680
  __shared__ float Qs[nHh * 30];     // 150
  __shared__ float Sc[nHh * Cc];     // 150
  const int n = blockIdx.x / Ll;
  const int l = blockIdx.x % Ll;
  const int tid = threadIdx.x;
  for (int i = tid; i < Cc * Dd; i += 256)
    X[i] = data[(size_t)n * (Tt * Dd) + (size_t)l * (Cc * Dd) + i];
  __syncthreads();
  // Q[h][t] = sum_{c,d} X[c,d] * Wq[h][c][d][t]
  if (tid < nHh * 30) {
    int h = tid / 30, t = tid % 30;
    const float* wq = Wq + (size_t)h * (Cc * Dd * 30) + t;
    float a = 0.f;
    for (int cd = 0; cd < Cc * Dd; ++cd) a += X[cd] * wq[(size_t)cd * 30];
    Qs[tid] = a;
  }
  __syncthreads();
  // QK[h][c] = sum_t (sum_d X[c,d]*Wk[h][d][t]) * Q[h][t] / sqrt(30)
  if (tid < nHh * Cc) {
    int h = tid / Cc, c = tid % Cc;
    float accT[30];
#pragma unroll
    for (int t = 0; t < 30; ++t) accT[t] = 0.f;
    const float* wk = Wk + (size_t)h * (Dd * 30);
    for (int d = 0; d < Dd; ++d) {
      float xd = X[c * Dd + d];
      const float* wr = wk + d * 30;
#pragma unroll
      for (int t = 0; t < 30; ++t) accT[t] += xd * wr[t];
    }
    float a = 0.f;
    const float* q = Qs + h * 30;
#pragma unroll
    for (int t = 0; t < 30; ++t) a += accT[t] * q[t];
    Sc[tid] = a * 0.18257418583505536f;  // 1/sqrt(30)
  }
  __syncthreads();
  if (tid < nHh) {
    float* s = Sc + tid * Cc;
    float mx = -1e30f;
    for (int c = 0; c < Cc; ++c) mx = fmaxf(mx, s[c]);
    float sum = 0.f;
    for (int c = 0; c < Cc; ++c) { float e = __expf(s[c] - mx); s[c] = e; sum += e; }
    float inv = 1.0f / sum;
    for (int c = 0; c < Cc; ++c) s[c] *= inv;
  }
  __syncthreads();
  for (int task = tid; task < nHh * Dd; task += 256) {
    int h = task / Dd, d = task % Dd;
    float a = 0.f;
    const float* s = Sc + h * Cc;
    for (int c = 0; c < Cc; ++c) a += s[c] * X[c * Dd + d];
    vals[((size_t)n * Ll + l) * Ff + h * Dd + d] = a;
  }
}

// ---------------- batchnorm stats -> affine ----------------
__global__ void __launch_bounds__(256)
bn_stats(const float* __restrict__ vals, const float* __restrict__ gamma,
         const float* __restrict__ beta, float* __restrict__ bn_a, float* __restrict__ bn_b)
{
  __shared__ float s1[256], s2[256];
  int f = blockIdx.x, tid = threadIdx.x;
  float sum = 0.f, ss = 0.f;
  for (int i = tid; i < Nn * Ll; i += 256) {
    float v = vals[(size_t)i * Ff + f];
    sum += v; ss += v * v;
  }
  s1[tid] = sum; s2[tid] = ss;
  __syncthreads();
  for (int st = 128; st > 0; st >>= 1) {
    if (tid < st) { s1[tid] += s1[tid + st]; s2[tid] += s2[tid + st]; }
    __syncthreads();
  }
  if (tid == 0) {
    float m = s1[0] / (float)(Nn * Ll);
    float var = s2[0] / (float)(Nn * Ll) - m * m;
    float a = gamma[f] * rsqrtf(var + EPSBN);
    bn_a[f] = a;
    bn_b[f] = beta[f] - m * a;
  }
}

// ---------------- build V = bn(vals)[b%N] + pertur/PERT ----------------
__global__ void build_v(const float* __restrict__ vals, const float* __restrict__ pert,
                        const float* __restrict__ bn_a, const float* __restrict__ bn_b,
                        float* __restrict__ V)
{
  size_t idx = (size_t)blockIdx.x * 256 + threadIdx.x;
  if (idx >= (size_t)Bb * LF) return;
  int b = (int)(idx / LF);
  int rem = (int)(idx % LF);
  int f = rem % Ff;
  V[idx] = vals[(size_t)(b % Nn) * LF + rem] * bn_a[f] + bn_b[f] + pert[idx] * (1.0f / PERT);
}

// ---------------- f32 -> bf16 weight staging (optional transpose, K zero-pad) ------
__global__ void cvt_bf16(const float* __restrict__ src, unsigned short* __restrict__ dst,
                         int rows, int cols, int Kpad, int transpose)
{
  size_t idx = (size_t)blockIdx.x * 256 + threadIdx.x;
  if (idx >= (size_t)rows * Kpad) return;
  int n = (int)(idx / Kpad), k = (int)(idx % Kpad);
  float v = 0.f;
  if (k < cols) v = transpose ? src[(size_t)k * rows + n] : src[(size_t)n * cols + k];
  dst[idx] = f2bf(v);
}

// ---------------- GRU gate kernels ----------------
__global__ void gate_fwd(const float* __restrict__ gi, const float* __restrict__ gh,
                         const float* __restrict__ bih, const float* __restrict__ bhh,
                         const float* __restrict__ Hprev, float* __restrict__ Hnew)
{
  int idx = blockIdx.x * 256 + threadIdx.x;
  if (idx >= Bb * Hh) return;
  int b = idx / Hh, j = idx % Hh;
  const float* gib = gi + (size_t)b * H3;
  const float* ghb = gh + (size_t)b * H3;
  float ir = gib[j] + bih[j], iz = gib[Hh + j] + bih[Hh + j], in_ = gib[2 * Hh + j] + bih[2 * Hh + j];
  float hr = ghb[j] + bhh[j], hz = ghb[Hh + j] + bhh[Hh + j], hn = ghb[2 * Hh + j] + bhh[2 * Hh + j];
  float r = sigm(ir + hr), z = sigm(iz + hz);
  float nn = tanhf(in_ + r * hn);
  Hnew[idx] = (1.f - z) * nn + z * Hprev[idx];
}

__global__ void gate_bwd(const float* __restrict__ gi, const float* __restrict__ gh,
                         const float* __restrict__ bih, const float* __restrict__ bhh,
                         const float* __restrict__ Hprev, const float* __restrict__ dh,
                         float* __restrict__ dgi, float* __restrict__ dgh,
                         float* __restrict__ dhtmp)
{
  int idx = blockIdx.x * 256 + threadIdx.x;
  if (idx >= Bb * Hh) return;
  int b = idx / Hh, j = idx % Hh;
  const float* gib = gi + (size_t)b * H3;
  const float* ghb = gh + (size_t)b * H3;
  float ir = gib[j] + bih[j], iz = gib[Hh + j] + bih[Hh + j], in_ = gib[2 * Hh + j] + bih[2 * Hh + j];
  float hr = ghb[j] + bhh[j], hz = ghb[Hh + j] + bhh[Hh + j], hn = ghb[2 * Hh + j] + bhh[2 * Hh + j];
  float r = sigm(ir + hr), z = sigm(iz + hz);
  float nn = tanhf(in_ + r * hn);
  float hp = Hprev[idx], dht = dh[idx];
  float dz = dht * (hp - nn);
  float dn = dht * (1.f - z);
  float dnp = dn * (1.f - nn * nn);
  float dr = dnp * hn;
  float dghn = dnp * r;
  float dzp = dz * z * (1.f - z);
  float drp = dr * r * (1.f - r);
  float* dgib = dgi + (size_t)b * H3;
  float* dghb = dgh + (size_t)b * H3;
  dgib[j] = drp; dgib[Hh + j] = dzp; dgib[2 * Hh + j] = dnp;
  dghb[j] = drp; dghb[Hh + j] = dzp; dghb[2 * Hh + j] = dghn;
  dhtmp[idx] = dht * z;
}

__global__ void dh_init(const float* __restrict__ Wout, float* __restrict__ dh)
{
  int idx = blockIdx.x * 256 + threadIdx.x;
  if (idx >= Bb * Hh) return;
  dh[idx] = Wout[idx % Hh];
}

// ---------------- outputs ----------------
__global__ void __launch_bounds__(256)
yh_kernel(const float* __restrict__ Wout, const float* __restrict__ bout,
          const float* __restrict__ HsL, float* __restrict__ out)
{
  __shared__ float s[256];
  int b = blockIdx.x, tid = threadIdx.x;
  s[tid] = Wout[tid] * HsL[(size_t)b * Hh + tid];
  __syncthreads();
  for (int st = 128; st > 0; st >>= 1) { if (tid < st) s[tid] += s[tid + st]; __syncthreads(); }
  if (tid == 0) out[1 + b] = sigm(s[0] + bout[0]);
}

__global__ void zero_acc(float* acc) { if (threadIdx.x < 4) acc[threadIdx.x] = 0.f; }

__global__ void fill_zero(float* __restrict__ p, size_t n)
{
  size_t idx = (size_t)blockIdx.x * 256 + threadIdx.x;
  if (idx < n) p[idx] = 0.f;
}

__global__ void __launch_bounds__(256)
sumsq(const float* __restrict__ p, int n, float* __restrict__ acc)
{
  __shared__ float s[256];
  int tid = threadIdx.x;
  float a = 0.f;
  for (size_t i = (size_t)blockIdx.x * 256 + tid; i < (size_t)n; i += (size_t)gridDim.x * 256) {
    float v = p[i]; a += v * v;
  }
  s[tid] = a; __syncthreads();
  for (int st = 128; st > 0; st >>= 1) { if (tid < st) s[tid] += s[tid + st]; __syncthreads(); }
  if (tid == 0) atomicAdd(acc, s[0]);
}

__global__ void __launch_bounds__(256)
loss_diff(const float* __restrict__ GX, const float* __restrict__ ap,
          const float* __restrict__ Wla2, const float* __restrict__ bla2,
          float* __restrict__ acc)
{
  __shared__ float s[256];
  size_t idx = (size_t)blockIdx.x * 256 + threadIdx.x;
  float contrib = 0.f;
  if (idx < (size_t)Bb * LF) {
    int b = (int)(idx / LF), j = (int)(idx % LF);
    const float* apr = ap + (size_t)b * 56;
    const float* wr  = Wla2 + (size_t)j * 56;
    float v = bla2[j];
#pragma unroll 8
    for (int k = 0; k < 56; ++k) v += apr[k] * wr[k];
    float d = GX[idx] - v;
    contrib = d * d;
  }
  s[threadIdx.x] = contrib; __syncthreads();
  for (int st = 128; st > 0; st >>= 1) { if (threadIdx.x < st) s[threadIdx.x] += s[threadIdx.x + st]; __syncthreads(); }
  if (threadIdx.x == 0) atomicAdd(acc, s[0]);
}

__global__ void finalize(const float* __restrict__ acc, float* __restrict__ out)
{
  out[0] = sqrtf(acc[0]) + LAMB * sqrtf(acc[1]) * sqrtf(acc[2]);
}

// ---------------- host side ----------------
extern "C" void kernel_launch(void* const* d_in, const int* in_sizes, int n_in,
                              void* d_out, int out_size, void* d_ws, size_t ws_size,
                              hipStream_t stream)
{
  (void)in_sizes; (void)n_in; (void)out_size; (void)ws_size;
  const float* data  = (const float*)d_in[0];
  // d_in[1] = labels (unused by the reference computation)
  const float* pert  = (const float*)d_in[2];
  const float* Wq    = (const float*)d_in[3];
  const float* Wk    = (const float*)d_in[4];
  const float* gamma = (const float*)d_in[5];
  const float* beta  = (const float*)d_in[6];
  const float* Wih   = (const float*)d_in[7];
  const float* Whh   = (const float*)d_in[8];
  const float* bih   = (const float*)d_in[9];
  const float* bhh   = (const float*)d_in[10];
  const float* Wout  = (const float*)d_in[11];
  const float* bout  = (const float*)d_in[12];
  const float* Wla1  = (const float*)d_in[13];
  const float* bla1  = (const float*)d_in[14];
  const float* Wla2  = (const float*)d_in[15];
  const float* bla2  = (const float*)d_in[16];
  float* out = (float*)d_out;

  char* ws = (char*)d_ws;
  size_t off = 0;
  auto alloc = [&](size_t bytes) -> char* {
    char* p = ws + off;
    off += (bytes + 255) & ~(size_t)255;
    return p;
  };
  float* vals  = (float*)alloc((size_t)Nn * LF * 4);            // 11.5 MB
  float* bn_a  = (float*)alloc((size_t)Ff * 4);
  float* bn_b  = (float*)alloc((size_t)Ff * 4);
  float* V     = (float*)alloc((size_t)Bb * LF * 4);            // 114.7 MB
  float* Hs    = (float*)alloc((size_t)(Ll + 1) * Bb * Hh * 4); // 110.1 MB
  float* gi    = (float*)alloc((size_t)Bb * H3 * 4);            // 15.7 MB
  float* gh    = (float*)alloc((size_t)Bb * H3 * 4);
  float* dgi   = (float*)alloc((size_t)Bb * H3 * 4);
  float* dgh   = (float*)alloc((size_t)Bb * H3 * 4);
  float* dh    = (float*)alloc((size_t)Bb * Hh * 4);
  float* dhtmp = (float*)alloc((size_t)Bb * Hh * 4);
  float* GX    = (float*)alloc((size_t)Bb * LF * 4);            // 114.7 MB
  float* apb   = (float*)alloc((size_t)Bb * 56 * 4);
  float* acc   = (float*)alloc(4 * 4);
  unsigned short* Wih_bt  = (unsigned short*)alloc((size_t)768 * 288 * 2);
  unsigned short* Whh_bt  = (unsigned short*)alloc((size_t)768 * 256 * 2);
  unsigned short* WhhT_bt = (unsigned short*)alloc((size_t)256 * 768 * 2);
  unsigned short* WihT_bt = (unsigned short*)alloc((size_t)280 * 768 * 2);
  unsigned short* Wla1_bt = (unsigned short*)alloc((size_t)56 * 5600 * 2);

  auto cdiv = [](long long a, long long b) { return (int)((a + b - 1) / b); };
  const size_t BH = (size_t)Bb * Hh;

  zero_acc<<<1, 64, 0, stream>>>(acc);
  attn_kernel<<<Nn * Ll, 256, 0, stream>>>(data, Wq, Wk, vals);
  bn_stats<<<Ff, 256, 0, stream>>>(vals, gamma, beta, bn_a, bn_b);
  build_v<<<cdiv((long long)Bb * LF, 256), 256, 0, stream>>>(vals, pert, bn_a, bn_b, V);
  cvt_bf16<<<cdiv(768 * 288, 256), 256, 0, stream>>>(Wih, Wih_bt, 768, 280, 288, 0);
  cvt_bf16<<<cdiv(768 * 256, 256), 256, 0, stream>>>(Whh, Whh_bt, 768, 256, 256, 0);
  cvt_bf16<<<cdiv(256 * 768, 256), 256, 0, stream>>>(Whh, WhhT_bt, 256, 768, 768, 1);
  cvt_bf16<<<cdiv(280 * 768, 256), 256, 0, stream>>>(Wih, WihT_bt, 280, 768, 768, 1);
  cvt_bf16<<<cdiv(56 * 5600, 256), 256, 0, stream>>>(Wla1, Wla1_bt, 56, 5600, 5600, 0);
  fill_zero<<<cdiv((long long)BH, 256), 256, 0, stream>>>(Hs, BH);

  dim3 blk(128);
  // ---- GRU forward ----
  for (int t = 0; t < Ll; ++t) {
    gemm_bf16<<<dim3(Bb / 128, 768 / 64), blk, 0, stream>>>(
        V + (size_t)t * Ff, LF, Wih_bt, 288, gi, H3, nullptr, 0, nullptr, Bb, H3, Ff, 288);
    gemm_bf16<<<dim3(Bb / 128, 768 / 64), blk, 0, stream>>>(
        Hs + (size_t)t * BH, Hh, Whh_bt, 256, gh, H3, nullptr, 0, nullptr, Bb, H3, Hh, 256);
    gate_fwd<<<cdiv((long long)BH, 256), 256, 0, stream>>>(
        gi, gh, bih, bhh, Hs + (size_t)t * BH, Hs + (size_t)(t + 1) * BH);
  }
  yh_kernel<<<Bb, 256, 0, stream>>>(Wout, bout, Hs + (size_t)Ll * BH, out);

  // ---- GRU backward (recompute gates; dh_L = W_out) ----
  dh_init<<<cdiv((long long)BH, 256), 256, 0, stream>>>(Wout, dh);
  for (int t = Ll - 1; t >= 0; --t) {
    gemm_bf16<<<dim3(Bb / 128, 768 / 64), blk, 0, stream>>>(
        V + (size_t)t * Ff, LF, Wih_bt, 288, gi, H3, nullptr, 0, nullptr, Bb, H3, Ff, 288);
    gemm_bf16<<<dim3(Bb / 128, 768 / 64), blk, 0, stream>>>(
        Hs + (size_t)t * BH, Hh, Whh_bt, 256, gh, H3, nullptr, 0, nullptr, Bb, H3, Hh, 256);
    gate_bwd<<<cdiv((long long)BH, 256), 256, 0, stream>>>(
        gi, gh, bih, bhh, Hs + (size_t)t * BH, dh, dgi, dgh, dhtmp);
    gemm_bf16<<<dim3(Bb / 128, cdiv(Hh, 64)), blk, 0, stream>>>(
        dgh, H3, WhhT_bt, 768, dh, Hh, dhtmp, Hh, nullptr, Bb, Hh, H3, 768);
    gemm_bf16<<<dim3(Bb / 128, cdiv(Ff, 64)), blk, 0, stream>>>(
        dgi, H3, WihT_bt, 768, GX + (size_t)t * Ff, LF, nullptr, 0, nullptr, Bb, Ff, H3, 768);
  }

  // ---- ap = V @ W_la1^T + b_la1, then loss ----
  gemm_bf16<<<dim3(Bb / 128, 1), blk, 0, stream>>>(
      V, LF, Wla1_bt, 5600, apb, 56, nullptr, 0, bla1, Bb, 56, LF, 5600);
  sumsq<<<256, 256, 0, stream>>>(Wla1, 56 * 5600, acc + 1);
  sumsq<<<256, 256, 0, stream>>>(Wla2, 5600 * 56, acc + 2);
  loss_diff<<<cdiv((long long)Bb * LF, 256), 256, 0, stream>>>(GX, apb, Wla2, bla2, acc);
  finalize<<<1, 1, 0, stream>>>(acc, out);
}
